// RetrieverPolyencoder_double_83932250898539
// MI455X (gfx1250) — matrix operand order repaired and checked by reading, not compiled, not run on hardware
//
#include <hip/hip_runtime.h>
#include <hip/hip_bf16.h>

// ---------------------------------------------------------------------------
// Poly-encoder scorer for MI455X (gfx1250), bf16 WMMA pipeline.
//  B=128, S=300, C=64, H=768, M=300.
//  ALL row/N/K side dims padded to 320 (multiple of 16 and 32) so every WMMA
//  loop is guard-free (no exec-mask churn); padding is exact:
//    - padded source rows are zero (cvt_pad / transpose write zeros)
//    - softmax gives exact 0 probability on padded columns (logit = -inf)
//    - w2 columns >= 300 stored as zero, so GEMM4's K-padding contributes 0
// ---------------------------------------------------------------------------

typedef __attribute__((ext_vector_type(16))) __bf16 bf16x16;
typedef __attribute__((ext_vector_type(8)))  float  floatx8;

union Frag16 {
    bf16x16        v;
    unsigned short h[16];
    unsigned int   u[8];
};

__device__ __forceinline__ unsigned short f2bf(float f) {
    union { float f; unsigned int u; } x; x.f = f;
    unsigned int r = x.u + 0x7FFFu + ((x.u >> 16) & 1u);   // round-to-nearest-even
    return (unsigned short)(r >> 16);
}

__device__ __forceinline__ unsigned int f2bf_pk(float lo, float hi) {
    return (unsigned int)f2bf(lo) | ((unsigned int)f2bf(hi) << 16);
}

__device__ __forceinline__ floatx8 wmma_bf16(bf16x16 a, bf16x16 b, floatx8 c) {
    return __builtin_amdgcn_wmma_f32_16x16x32_bf16(false, a, false, b, (short)0, c, false, false);
}

// A-fragment (16x32 bf16): lane L -> row M=L&15; VGPR v: K = k0 + 16h + 2v.
// Lowered to 2x global_load_b128 (bytes [0,32) from kb).
__device__ __forceinline__ bf16x16 load_a16(const unsigned short* rowPtr, int k0, int half) {
    Frag16 f;
    int kb = k0 + half * 16;
#pragma unroll
    for (int v = 0; v < 4; ++v) f.u[v]     = *(const unsigned int*)(rowPtr + kb + 2 * v);
#pragma unroll
    for (int v = 0; v < 4; ++v) f.u[4 + v] = *(const unsigned int*)(rowPtr + kb + 8 + 2 * v);
    return f.v;
}

// B-fragment (32x16 bf16): lane L -> col N=L&15; VGPR v: K = k0 + 16h + 2v.
// 32 contiguous bytes -> 2x global_load_b128.
__device__ __forceinline__ bf16x16 load_b16(const unsigned short* colPtr, int k0, int half) {
    Frag16 f;
    int kb = k0 + half * 16;
#pragma unroll
    for (int v = 0; v < 8; ++v) f.u[v] = *(const unsigned int*)(colPtr + kb + 2 * v);
    return f.v;
}

// ---------------------------------------------------------------------------
// 1) fp32 -> bf16 with zero row padding:  dst[b][r][i] = r<valid ? src : 0
// ---------------------------------------------------------------------------
__global__ void cvt_pad_kernel(const float* __restrict__ src, unsigned short* __restrict__ dst,
                               int valid, int pad, int inner, long total) {
    for (long i = blockIdx.x * (long)blockDim.x + threadIdx.x; i < total;
         i += (long)gridDim.x * blockDim.x) {
        long rem = i;
        int  c = (int)(rem % inner); rem /= inner;
        int  r = (int)(rem % pad);
        long b = rem / pad;
        float v = 0.f;
        if (r < valid) v = src[(b * valid + r) * (long)inner + c];
        dst[i] = f2bf(v);
    }
}

// ---------------------------------------------------------------------------
// 2) ceT[b][h][s320] = ce[b][s][h] (bf16), zero for s>=300.  LDS-tiled 32x32.
// ---------------------------------------------------------------------------
__global__ __launch_bounds__(256) void transpose_kernel(const float* __restrict__ ce,
                                                        unsigned short* __restrict__ cet) {
    __shared__ float tile[32][33];
    int hx = blockIdx.x, sy = blockIdx.y, b = blockIdx.z;
    int tx = threadIdx.x, ty = threadIdx.y;
    int h = hx * 32 + tx;
#pragma unroll
    for (int r = 0; r < 4; ++r) {
        int s = sy * 32 + ty + r * 8;
        float v = 0.f;
        if (s < 300) v = ce[((size_t)b * 300 + s) * 768 + h];
        tile[ty + r * 8][tx] = v;
    }
    __syncthreads();
#pragma unroll
    for (int r = 0; r < 4; ++r) {
        int h2 = hx * 32 + ty + r * 8;
        int s2 = sy * 32 + tx;
        cet[((size_t)b * 768 + h2) * 320 + s2] = f2bf(tile[tx][ty + r * 8]);
    }
}

// ---------------------------------------------------------------------------
// 3) Fused GEMM1 + mask + softmax.
//    logits[16 x 320] = pos[m-tile] @ ce[b]^T, *mask, softmax over s -> P bf16 [B][320][320].
//    Block: (b, m-tile 0..19).  4 waves x 5 s-tiles = 20 tiles, guard-free.
//    K = 768 (24 WMMA steps).  Padded s-columns get logit=-inf -> prob exactly 0.
// ---------------------------------------------------------------------------
__global__ __launch_bounds__(128) void gemm1_softmax_kernel(
        const unsigned short* __restrict__ posb,   // [320][768]
        const unsigned short* __restrict__ ceb,    // [B][320][768]
        const int*            __restrict__ mask,   // [B][300]
        unsigned short*       __restrict__ Pb) {   // [B][320][320]
    __shared__ float lg[16][320];
    __shared__ float red[16][8];
    __shared__ float rowstat[16];

    int mt = blockIdx.x, b = blockIdx.y;
    int tid = threadIdx.x, w = tid >> 5, lane = tid & 31;
    int half = lane >> 4, l16 = lane & 15;

    const unsigned short* aRow = posb + (size_t)(mt * 16 + l16) * 768;
    const unsigned short* bRow[5];
#pragma unroll
    for (int j = 0; j < 5; ++j) {
        int s = (w + 4 * j) * 16 + l16;                 // nt = w + 4j covers 0..19
        bRow[j] = ceb + ((size_t)b * 320 + s) * 768;
    }

    floatx8 acc[5] = {{}, {}, {}, {}, {}};

    for (int kk = 0; kk < 24; ++kk) {
        int k0 = kk * 32;
        bf16x16 a = load_a16(aRow, k0, half);
        bf16x16 bb[5];
#pragma unroll
        for (int j = 0; j < 5; ++j) bb[j] = load_b16(bRow[j], k0, half);
#pragma unroll
        for (int j = 0; j < 5; ++j) acc[j] = wmma_bf16(a, bb[j], acc[j]);
    }

    // multiplicative mask (masked zeros still participate in softmax); -inf on padded cols
#pragma unroll
    for (int j = 0; j < 5; ++j) {
        int s = (w + 4 * j) * 16 + l16;
        bool valid = (s < 300);
        int  sc = valid ? s : 299;                      // clamped, always in-bounds
        float mv = (float)mask[b * 300 + sc];
#pragma unroll
        for (int v = 0; v < 8; ++v) {
            int r = v + 8 * half;
            lg[r][s] = valid ? acc[j][v] * mv : -__builtin_inff();
        }
    }
    __syncthreads();

    // block softmax: 8 threads per row, 40-column chunks (8*40 == 320)
    int r = tid >> 3, c8 = tid & 7;
    float mx = -__builtin_inff();
    for (int c = c8 * 40; c < c8 * 40 + 40; ++c) mx = fmaxf(mx, lg[r][c]);
    red[r][c8] = mx;
    __syncthreads();
    if (c8 == 0) {
        float m2 = red[r][0];
#pragma unroll
        for (int i = 1; i < 8; ++i) m2 = fmaxf(m2, red[r][i]);
        rowstat[r] = m2;
    }
    __syncthreads();
    float rm = rowstat[r];
    float sm = 0.f;
    for (int c = c8 * 40; c < c8 * 40 + 40; ++c) {
        float e = __expf(lg[r][c] - rm);
        lg[r][c] = e;
        sm += e;
    }
    __syncthreads();
    red[r][c8] = sm;
    __syncthreads();
    if (c8 == 0) {
        float s2 = 0.f;
#pragma unroll
        for (int i = 0; i < 8; ++i) s2 += red[r][i];
        rowstat[r] = s2;
    }
    __syncthreads();
    float inv = 1.f / rowstat[r];
    unsigned int* prow = (unsigned int*)(Pb + ((size_t)b * 320 + mt * 16 + r) * 320);
    for (int c = c8 * 20; c < c8 * 20 + 20; ++c)        // packed dword stores (2 cols each)
        prow[c] = f2bf_pk(lg[r][2 * c] * inv, lg[r][2 * c + 1] * inv);
}

// ---------------------------------------------------------------------------
// 4) GEMM2: att = P[b] (320x320) @ ce[b] via ceT layout (K=s, N=h).  K=320 (10 steps).
//    Block: (m-tile 0..19, h-group of 12 tiles, b); each wave -> 3 h-tiles.
//    Writes BOTH att[b][m320][h768] (GEMM3's B) and attT[b][h][m320] (GEMM4's B).
// ---------------------------------------------------------------------------
__global__ __launch_bounds__(128) void gemm2_kernel(
        const unsigned short* __restrict__ Pb,     // [B][320][320]
        const unsigned short* __restrict__ cetb,   // [B][768][320]
        unsigned short*       __restrict__ attb,   // [B][320][768]
        unsigned short*       __restrict__ attTb) {// [B][768][320]
    int mt = blockIdx.x, hg = blockIdx.y, b = blockIdx.z;
    int tid = threadIdx.x, w = tid >> 5, lane = tid & 31;
    int half = lane >> 4, l16 = lane & 15;

    const unsigned short* aRow = Pb + ((size_t)b * 320 + mt * 16 + l16) * 320;
    int htBase = hg * 12 + w * 3;
    const unsigned short* bRow[3];
#pragma unroll
    for (int j = 0; j < 3; ++j)
        bRow[j] = cetb + ((size_t)b * 768 + (htBase + j) * 16 + l16) * 320;

    floatx8 acc[3] = {{}, {}, {}};
    for (int kk = 0; kk < 10; ++kk) {
        int k0 = kk * 32;
        bf16x16 a = load_a16(aRow, k0, half);
        bf16x16 bb[3];
#pragma unroll
        for (int j = 0; j < 3; ++j) bb[j] = load_b16(bRow[j], k0, half);
#pragma unroll
        for (int j = 0; j < 3; ++j) acc[j] = wmma_bf16(a, bb[j], acc[j]);
    }
#pragma unroll
    for (int j = 0; j < 3; ++j) {
        int h = (htBase + j) * 16 + l16;
        int m0 = mt * 16 + 8 * half;
        // att[m][h]: rows stride 768 -> 8 b16 stores
#pragma unroll
        for (int v = 0; v < 8; ++v)
            attb[((size_t)b * 320 + m0 + v) * 768 + h] = f2bf(acc[j][v]);
        // attT[h][m]: m contiguous -> 4 packed dword stores
        unsigned int* trow = (unsigned int*)(attTb + ((size_t)b * 768 + h) * 320 + m0);
#pragma unroll
        for (int v = 0; v < 4; ++v)
            trow[v] = f2bf_pk(acc[j][2 * v], acc[j][2 * v + 1]);
    }
}

// ---------------------------------------------------------------------------
// 5) GEMM3: w2[b][64][320] = resp[b] (64x768) @ att[b]^T  (K=h, N=m).  K=768 (24 steps).
//    Block: (m-tile 0..19, b); wave = c-tile.  Cols m>=300 stored as zero (exact K-pad).
// ---------------------------------------------------------------------------
__global__ __launch_bounds__(128) void gemm3_kernel(
        const unsigned short* __restrict__ respb, // [B][64][768]
        const unsigned short* __restrict__ attb,  // [B][320][768]
        unsigned short*       __restrict__ w2b) { // [B][64][320]
    int nt = blockIdx.x, b = blockIdx.y;
    int tid = threadIdx.x, w = tid >> 5, lane = tid & 31;
    int half = lane >> 4, l16 = lane & 15;

    int c    = w * 16 + l16;
    int mcol = nt * 16 + l16;
    const unsigned short* aRow = respb + ((size_t)b * 64 + c) * 768;
    const unsigned short* bRow = attb + ((size_t)b * 320 + mcol) * 768;

    floatx8 acc = {};
    for (int kk = 0; kk < 24; ++kk) {
        int k0 = kk * 32;
        bf16x16 a  = load_a16(aRow, k0, half);
        bf16x16 bb = load_b16(bRow, k0, half);
        acc = wmma_bf16(a, bb, acc);
    }
#pragma unroll
    for (int v = 0; v < 8; ++v) {
        int cc  = w * 16 + v + 8 * half;
        float val = (mcol < 300) ? acc[v] : 0.f;    // exact zero K-padding for GEMM4
        w2b[((size_t)b * 64 + cc) * 320 + mcol] = f2bf(val);
    }
}

// ---------------------------------------------------------------------------
// 6) Fused GEMM4 + dot: emb = w2 (64x320) @ att (K=m, N=h) via attT; out = sum_h emb*resp.
//    Block: (c-tile 0..3, b).  Each wave sweeps 12 h-tiles; emb never leaves registers.
//    Reduction: butterfly shuffles within 16-lane halves + LDS across waves.
// ---------------------------------------------------------------------------
__global__ __launch_bounds__(128) void gemm4_dot_kernel(
        const unsigned short* __restrict__ w2b,   // [B][64][320]
        const unsigned short* __restrict__ attTb, // [B][768][320]
        const float*          __restrict__ resp,  // [B][64][768] fp32
        float*                __restrict__ out) { // [B][64]
    __shared__ float dotbuf[4][16];
    int ct = blockIdx.x, b = blockIdx.y;
    int tid = threadIdx.x, w = tid >> 5, lane = tid & 31;
    int half = lane >> 4, l16 = lane & 15;

    const unsigned short* aRow = w2b + ((size_t)b * 64 + ct * 16 + l16) * 320;
    const unsigned short* attT = attTb + (size_t)b * 768 * 320;

    // hoist all A fragments (K=320 -> 10 fragments), reused across 12 h-tiles
    bf16x16 aF[10];
#pragma unroll
    for (int kk = 0; kk < 10; ++kk) aF[kk] = load_a16(aRow, kk * 32, half);

    float pd[8] = {0.f, 0.f, 0.f, 0.f, 0.f, 0.f, 0.f, 0.f};

    for (int j = 0; j < 12; ++j) {
        int h = (w * 12 + j) * 16 + l16;
        const unsigned short* bRow = attT + (size_t)h * 320;
        floatx8 acc = {};
#pragma unroll
        for (int kk = 0; kk < 10; ++kk)
            acc = wmma_bf16(aF[kk], load_b16(bRow, kk * 32, half), acc);
#pragma unroll
        for (int v = 0; v < 8; ++v) {
            int cc = ct * 16 + v + 8 * half;
            float rv = resp[((size_t)b * 64 + cc) * 768 + h];
            pd[v] += acc[v] * rv;
        }
    }

    // reduce over the 16 N-lanes of each half
#pragma unroll
    for (int v = 0; v < 8; ++v) {
        float x = pd[v];
#pragma unroll
        for (int msk = 8; msk >= 1; msk >>= 1) x += __shfl_xor(x, msk, 16);
        pd[v] = x;
    }
    if (l16 == 0) {
#pragma unroll
        for (int v = 0; v < 8; ++v) dotbuf[w][half * 8 + v] = pd[v];
    }
    __syncthreads();
    if (tid < 16) {
        float s = dotbuf[0][tid] + dotbuf[1][tid] + dotbuf[2][tid] + dotbuf[3][tid];
        out[(size_t)b * 64 + ct * 16 + tid] = s;
    }
}

// ---------------------------------------------------------------------------
// Launch.  Inputs: [0]=context_encoded f32 [128,300,768], [1]=responses f32 [128,64,768],
//                  [2]=pos_emb f32 [300,768], [3]=context_mask i32 [128,300].
// Output: f32 [128,64].  Workspace: ~296 MB of bf16 staging in d_ws.
// ---------------------------------------------------------------------------
extern "C" void kernel_launch(void* const* d_in, const int* in_sizes, int n_in,
                              void* d_out, int out_size, void* d_ws, size_t ws_size,
                              hipStream_t stream) {
    (void)in_sizes; (void)n_in; (void)out_size; (void)ws_size;
    const float* ce   = (const float*)d_in[0];
    const float* resp = (const float*)d_in[1];
    const float* pos  = (const float*)d_in[2];
    const int*   mask = (const int*)d_in[3];
    float*       out  = (float*)d_out;

    unsigned short* ws = (unsigned short*)d_ws;
    size_t o = 0;
    unsigned short* posb  = ws + o; o += 320UL * 768;             //  0.47 MiB
    unsigned short* ceb   = ws + o; o += 128UL * 320 * 768;       // 60.0  MiB
    unsigned short* respb = ws + o; o += 128UL * 64 * 768;        // 12.0  MiB
    unsigned short* cetb  = ws + o; o += 128UL * 768 * 320;       // 60.0  MiB
    unsigned short* Pb    = ws + o; o += 128UL * 320 * 320;       // 25.0  MiB
    unsigned short* attb  = ws + o; o += 128UL * 320 * 768;       // 60.0  MiB
    unsigned short* attTb = ws + o; o += 128UL * 768 * 320;       // 60.0  MiB
    unsigned short* w2b   = ws + o; o += 128UL * 64 * 320;        //  5.0  MiB

    long tpos  = 320L * 768;
    long tce   = 128L * 320 * 768;
    long tresp = 128L * 64 * 768;
    cvt_pad_kernel<<<dim3((unsigned)((tpos + 255) / 256)), 256, 0, stream>>>(pos, posb, 300, 320, 768, tpos);
    cvt_pad_kernel<<<dim3(4096), 256, 0, stream>>>(ce, ceb, 300, 320, 768, tce);
    cvt_pad_kernel<<<dim3(2048), 256, 0, stream>>>(resp, respb, 64, 64, 768, tresp);
    transpose_kernel<<<dim3(24, 10, 128), dim3(32, 8), 0, stream>>>(ce, cetb);

    gemm1_softmax_kernel<<<dim3(20, 128), 128, 0, stream>>>(posb, ceb, mask, Pb);
    gemm2_kernel<<<dim3(20, 4, 128), 128, 0, stream>>>(Pb, cetb, attb, attTb);
    gemm3_kernel<<<dim3(20, 128), 128, 0, stream>>>(respb, attb, w2b);
    gemm4_dot_kernel<<<dim3(4, 128), 128, 0, stream>>>(w2b, attTb, resp, out);
}
